// SparseAutoencoder_62423054680299
// MI455X (gfx1250) — compile-verified, matrix operand order checked by reading, not run on hardware
//
#include <hip/hip_runtime.h>
#include <hip/hip_bf16.h>
#include <stdint.h>

// ---------------- problem constants ----------------
constexpr int BATCH = 4096;
constexpr int DIM   = 2048;   // D
constexpr int NEUR  = 16384;  // M
constexpr int TOPK  = 64;
constexpr int MULTIK = 256;
constexpr int AUXK  = 128;
constexpr int DEAD_THRESH = 256;

typedef unsigned int u32;
typedef unsigned long long u64;
typedef __attribute__((ext_vector_type(4)))  u32    u32x4;
typedef __attribute__((ext_vector_type(16))) __bf16 v16bf;
typedef __attribute__((ext_vector_type(8)))  float  v8f;

union Frag { v16bf v; u32x4 q[2]; };

// ---------------- workspace layout (bytes) ----------------
constexpr size_t OFF_A    = 0;                                   // bf16 xc  [B,D]
constexpr size_t SZ_A     = (size_t)BATCH * DIM * 2;
constexpr size_t OFF_W    = OFF_A + SZ_A;                        // bf16 Wenc[M,D]
constexpr size_t SZ_W     = (size_t)NEUR * DIM * 2;
constexpr size_t OFF_PRE  = OFF_W + SZ_W;                        // f32 pre  [B,M]
constexpr size_t SZ_PRE   = (size_t)BATCH * NEUR * 4;
constexpr size_t OFF_WT   = OFF_PRE + SZ_PRE;                    // f32 WdecT[M,D]
constexpr size_t SZ_WT    = (size_t)NEUR * DIM * 4;
constexpr size_t OFF_MKV  = OFF_WT + SZ_WT;                      // f32 multik vals [B,256]
constexpr size_t SZ_MKV   = (size_t)BATCH * MULTIK * 4;
constexpr size_t OFF_MKI  = OFF_MKV + SZ_MKV;                    // i32 multik idx
constexpr size_t OFF_TKV  = OFF_MKI + SZ_MKV;                    // f32 topk vals [B,64]
constexpr size_t SZ_TKV   = (size_t)BATCH * TOPK * 4;
constexpr size_t OFF_TKI  = OFF_TKV + SZ_TKV;                    // i32 topk idx
constexpr size_t OFF_ST   = OFF_TKI + SZ_TKV;                    // i32 steps2 [M]

// ---------------- conversion kernels ----------------
__global__ void k_convert_x(const float* __restrict__ x,
                            const float* __restrict__ input_bias,
                            __bf16* __restrict__ A) {
  size_t i = ((size_t)blockIdx.x * 256 + threadIdx.x) * 8;
#pragma unroll
  for (int e = 0; e < 8; ++e) {
    size_t j = i + e;
    float v = x[j] - input_bias[j & (DIM - 1)];
    A[j] = (__bf16)v;
  }
}

__global__ void k_convert_w(const float* __restrict__ W,
                            __bf16* __restrict__ Wb) {
  size_t i = ((size_t)blockIdx.x * 256 + threadIdx.x) * 8;
#pragma unroll
  for (int e = 0; e < 8; ++e) Wb[i + e] = (__bf16)W[i + e];
}

// W_dec [D,M] -> Wt [M,D]
__global__ void k_transpose(const float* __restrict__ Wdec, float* __restrict__ Wt) {
  __shared__ float tile[32][33];
  int m0 = blockIdx.x * 32, d0 = blockIdx.y * 32;
  int tx = threadIdx.x, ty = threadIdx.y;
#pragma unroll
  for (int j = 0; j < 32; j += 8)
    tile[ty + j][tx] = Wdec[(size_t)(d0 + ty + j) * NEUR + m0 + tx];
  __syncthreads();
#pragma unroll
  for (int j = 0; j < 32; j += 8)
    Wt[(size_t)(m0 + ty + j) * DIM + d0 + tx] = tile[tx][ty + j];
}

// ---------------- WMMA GEMM: pre[b,n] = sum_d A[b,d]*Wenc[n,d] + nbias[n] ----------------
// 256 threads (8 waves), tile 128(batch) x 128(neuron), K-slice 32,
// double-buffered LDS filled by GLOBAL_LOAD_ASYNC_TO_LDS_B128 (ASYNCcnt).
__global__ __launch_bounds__(256) void k_gemm(const __bf16* __restrict__ A,
                                              const __bf16* __restrict__ Wb,
                                              const float* __restrict__ nbias,
                                              float* __restrict__ pre) {
  __shared__ __align__(16) __bf16 As[2][128][40];  // +8 pad -> 80B rows
  __shared__ __align__(16) __bf16 Bs[2][128][40];
  constexpr unsigned BUFSZ = 128 * 40 * 2;         // 10240 B per buffer

  const int tid  = threadIdx.x;
  const int lane = tid & 31;
  const int w    = tid >> 5;
  const int wr   = w >> 2;   // 0..1  (batch-dir, 64 rows each)
  const int wc   = w & 3;    // 0..3  (neuron-dir, 32 cols each)
  const int n0   = blockIdx.x * 128;
  const int m0   = blockIdx.y * 128;

  // chunk assignment: 512 chunks of 8 bf16 (16B); thread owns chunk tid and tid+256
  const int r0 = tid >> 2, c0 = (tid & 3) << 3;
  const int r1 = r0 + 64;
  const u64 gA0 = (u64)(A  + (size_t)(m0 + r0) * DIM + c0);
  const u64 gA1 = (u64)(A  + (size_t)(m0 + r1) * DIM + c0);
  const u64 gB0 = (u64)(Wb + (size_t)(n0 + r0) * DIM + c0);
  const u64 gB1 = (u64)(Wb + (size_t)(n0 + r1) * DIM + c0);
  // workgroup-relative LDS byte offsets (flat LDS addr carries offset in [31:0])
  const unsigned lA0 = (unsigned)(u64)&As[0][r0][c0];
  const unsigned lA1 = (unsigned)(u64)&As[0][r1][c0];
  const unsigned lB0 = (unsigned)(u64)&Bs[0][r0][c0];
  const unsigned lB1 = (unsigned)(u64)&Bs[0][r1][c0];

  auto issue = [&](int stage, int buf) {
    const u64 go = (u64)stage * 64u;          // 32 bf16 per K-slice
    const unsigned lo = (unsigned)buf * BUFSZ;
    asm volatile("global_load_async_to_lds_b128 %0, %1, off"
                 :: "v"(lA0 + lo), "v"(gA0 + go) : "memory");
    asm volatile("global_load_async_to_lds_b128 %0, %1, off"
                 :: "v"(lA1 + lo), "v"(gA1 + go) : "memory");
    asm volatile("global_load_async_to_lds_b128 %0, %1, off"
                 :: "v"(lB0 + lo), "v"(gB0 + go) : "memory");
    asm volatile("global_load_async_to_lds_b128 %0, %1, off"
                 :: "v"(lB1 + lo), "v"(gB1 + go) : "memory");
  };

  v8f acc[4][2];
#pragma unroll
  for (int i = 0; i < 4; ++i)
#pragma unroll
    for (int j = 0; j < 2; ++j)
      acc[i][j] = (v8f){0.f,0.f,0.f,0.f,0.f,0.f,0.f,0.f};

  const int mrow = lane & 15;
  const int koff = (lane < 16) ? 0 : 8;   // ISA 16-bit A/B fragment layout

  constexpr int NSTAGE = DIM / 32;        // 64
  issue(0, 0);
  int buf = 0;
  for (int s = 0; s < NSTAGE; ++s) {
    asm volatile("s_wait_asynccnt 0x0" ::: "memory");  // stage-s tile landed (this wave)
    __syncthreads();                                   // all waves: data ready, prev buffer free
    if (s + 1 < NSTAGE) issue(s + 1, buf ^ 1);         // prefetch next slice (overlaps WMMA)

    const __bf16 (*Asb)[40] = As[buf];
    const __bf16 (*Bsb)[40] = Bs[buf];
    Frag fa[4], fb[2];
#pragma unroll
    for (int mt = 0; mt < 4; ++mt) {
      int r = wr * 64 + mt * 16 + mrow;
      fa[mt].q[0] = *(const u32x4*)&Asb[r][koff];
      fa[mt].q[1] = *(const u32x4*)&Asb[r][koff + 16];
    }
#pragma unroll
    for (int nt = 0; nt < 2; ++nt) {
      int r = wc * 32 + nt * 16 + mrow;
      fb[nt].q[0] = *(const u32x4*)&Bsb[r][koff];
      fb[nt].q[1] = *(const u32x4*)&Bsb[r][koff + 16];
    }
#pragma unroll
    for (int mt = 0; mt < 4; ++mt)
#pragma unroll
      for (int nt = 0; nt < 2; ++nt)
        acc[mt][nt] = __builtin_amdgcn_wmma_f32_16x16x32_bf16(
            false, fa[mt].v, false, fb[nt].v, (short)0, acc[mt][nt], false, false);
    buf ^= 1;
  }

  // epilogue: C layout — VGPR r holds row r (lanes 0-15) / r+8 (lanes 16-31), col = lane&15
  const int ncol  = n0 + wc * 32 + (lane & 15);
  const int rbase = m0 + wr * 64 + ((lane < 16) ? 0 : 8);
#pragma unroll
  for (int mt = 0; mt < 4; ++mt)
#pragma unroll
    for (int nt = 0; nt < 2; ++nt) {
      int nn = ncol + nt * 16;
      float nb = nbias[nn];
#pragma unroll
      for (int r = 0; r < 8; ++r)
        pre[(size_t)(rbase + mt * 16 + r) * NEUR + nn] = acc[mt][nt][r] + nb;
    }
}

// ---------------- per-row top-SEL radix select + bitonic sort ----------------
__device__ __forceinline__ u32 f2key(float f) {
  u32 u = __float_as_uint(f);
  return u ^ ((u >> 31) ? 0xFFFFFFFFu : 0x80000000u);  // monotonic: larger f -> larger key
}

__device__ __forceinline__ float fetch_val(const float* __restrict__ pre_row,
                                           const int* __restrict__ steps2, int i) {
  float v = pre_row[i];
  if (steps2) v = (steps2[i] > DEAD_THRESH) ? v : 0.0f;
  return v;
}

// blockDim.x == 256. out_val/out_idx are shared arrays of SEL entries; sorted desc on return.
template <int SEL>
__device__ void select_topk(const float* __restrict__ pre_row,
                            const int* __restrict__ steps2,
                            float* __restrict__ out_val,
                            int* __restrict__ out_idx) {
  __shared__ u32 hist[256];
  __shared__ int eq_idx[256];
  __shared__ u32 sh_pref, sh_mask;
  __shared__ int sh_need, sh_cgt, sh_ceq;
  const int tid = threadIdx.x;

  if (tid == 0) { sh_pref = 0u; sh_mask = 0u; sh_need = SEL; }
  for (int p = 3; p >= 0; --p) {
    hist[tid] = 0u;
    __syncthreads();
    u32 pref = sh_pref, mask = sh_mask;
    for (int i = tid; i < NEUR; i += 256) {
      u32 k = f2key(fetch_val(pre_row, steps2, i));
      if ((k & mask) == pref) atomicAdd(&hist[(k >> (8 * p)) & 255], 1u);
    }
    __syncthreads();
    if (tid == 0) {
      int need = sh_need; u32 cum = 0; int t = 0;
      for (int b = 255; b >= 0; --b) {
        u32 h = hist[b];
        if (cum + h >= (u32)need) { t = b; sh_need = need - (int)cum; break; }
        cum += h;
      }
      sh_pref = pref | ((u32)t << (8 * p));
      sh_mask = mask | (0xFFu << (8 * p));
    }
    __syncthreads();
  }
  const u32 T = sh_pref;
  const int r = sh_need;               // ties at threshold to take
  if (tid == 0) { sh_cgt = 0; sh_ceq = 0; }
  __syncthreads();

  for (int i = tid; i < NEUR; i += 256) {
    float v = fetch_val(pre_row, steps2, i);
    u32 k = f2key(v);
    if (k > T) {
      int pos = atomicAdd(&sh_cgt, 1);
      out_val[pos] = v; out_idx[pos] = i;
    } else if (k == T) {
      int p2 = atomicAdd(&sh_ceq, 1);
      if (p2 < 256) eq_idx[p2] = i;
    }
  }
  __syncthreads();
  // deterministic ties: sort candidate indices ascending, take r smallest
  int ec = sh_ceq > 256 ? 256 : sh_ceq;
  if (tid >= ec) eq_idx[tid] = 0x7FFFFFFF;
  for (int size = 2; size <= 256; size <<= 1)
    for (int stride = size >> 1; stride > 0; stride >>= 1) {
      __syncthreads();
      int j = tid ^ stride;
      if (j > tid) {
        bool up = ((tid & size) == 0);
        int a = eq_idx[tid], b = eq_idx[j];
        if ((a > b) == up) { eq_idx[tid] = b; eq_idx[j] = a; }
      }
    }
  __syncthreads();
  if (tid < r) {
    int ix = eq_idx[tid];
    out_idx[SEL - r + tid] = ix;
    out_val[SEL - r + tid] = fetch_val(pre_row, steps2, ix);
  }
  // bitonic sort SEL entries: (value desc, index asc)
  for (int size = 2; size <= SEL; size <<= 1)
    for (int stride = size >> 1; stride > 0; stride >>= 1) {
      __syncthreads();
      if (tid < SEL) {
        int j = tid ^ stride;
        if (j > tid) {
          float va = out_val[tid], vb = out_val[j];
          int ia = out_idx[tid], ib = out_idx[j];
          u32 ka = f2key(va), kb = f2key(vb);
          bool aFirst = (ka > kb) || (ka == kb && ia < ib);
          bool desc = ((tid & size) == 0);
          if (aFirst != desc) { out_val[tid] = vb; out_val[j] = va;
                                out_idx[tid] = ib; out_idx[j] = ia; }
        }
      }
    }
  __syncthreads();
}

// ---------------- top-k kernel: multik list, topk outputs, activations, steps scatter ----------------
__global__ __launch_bounds__(256) void k_topk(const float* __restrict__ pre,
                                              int* __restrict__ ws_tk_i, float* __restrict__ ws_tk_v,
                                              int* __restrict__ ws_mk_i, float* __restrict__ ws_mk_v,
                                              int* __restrict__ steps2,
                                              float* __restrict__ act,
                                              float* __restrict__ out_tki, float* __restrict__ out_tkv) {
  __shared__ float s_val[256];
  __shared__ int   s_idx[256];
  const int row = blockIdx.x, tid = threadIdx.x;
  const float* pre_row = pre + (size_t)row * NEUR;

  select_topk<MULTIK>(pre_row, nullptr, s_val, s_idx);

  float v  = s_val[tid];
  int   ix = s_idx[tid];
  float rv = v > 0.f ? v : 0.f;
  ws_mk_v[(size_t)row * MULTIK + tid] = rv;
  ws_mk_i[(size_t)row * MULTIK + tid] = ix;
  if (tid < TOPK) {
    out_tkv[(size_t)row * TOPK + tid] = rv;
    out_tki[(size_t)row * TOPK + tid] = (float)ix;
    ws_tk_v[(size_t)row * TOPK + tid] = rv;
    ws_tk_i[(size_t)row * TOPK + tid] = ix;
    steps2[ix] = 0;   // races across rows all write 0: benign & deterministic
  }
  // dense activations row: zero then scatter
  float* arow = act + (size_t)row * NEUR;
  for (int j = tid; j < NEUR; j += 256) arow[j] = 0.f;
  __syncthreads();
  if (tid < TOPK) arow[s_idx[tid]] = rv;
}

__global__ __launch_bounds__(256) void k_aux(const float* __restrict__ pre,
                                             const int* __restrict__ steps2,
                                             float* __restrict__ out_auxi,
                                             float* __restrict__ out_auxv) {
  __shared__ float s_val[AUXK];
  __shared__ int   s_idx[AUXK];
  const int row = blockIdx.x, tid = threadIdx.x;
  select_topk<AUXK>(pre + (size_t)row * NEUR, steps2, s_val, s_idx);
  if (tid < AUXK) {
    float v = s_val[tid];
    out_auxv[(size_t)row * AUXK + tid] = v > 0.f ? v : 0.f;
    out_auxi[(size_t)row * AUXK + tid] = (float)s_idx[tid];
  }
}

__global__ void k_steps_init(const int* __restrict__ steps, int* __restrict__ steps2) {
  int i = blockIdx.x * 256 + threadIdx.x;
  steps2[i] = steps[i] + 1;
}

// ---------------- sparse reconstruction: out[row,:] = bias + sum_j v_j * Wt[idx_j,:] ----------------
__global__ __launch_bounds__(256) void k_recon(const float* __restrict__ vals,
                                               const int* __restrict__ idxs,
                                               int count,
                                               const float* __restrict__ Wt,
                                               const float* __restrict__ input_bias,
                                               float* __restrict__ out) {
  __shared__ float sv[MULTIK];
  __shared__ int   si[MULTIK];
  const int row = blockIdx.x, tid = threadIdx.x;
  if (tid < count) {
    sv[tid] = vals[(size_t)row * count + tid];
    si[tid] = idxs[(size_t)row * count + tid];
  }
  __syncthreads();
  float acc[DIM / 256];
#pragma unroll
  for (int e = 0; e < DIM / 256; ++e) acc[e] = input_bias[tid + e * 256];
  for (int j = 0; j < count; ++j) {
    float v = sv[j];
    const float* wrow = Wt + (size_t)si[j] * DIM;
#pragma unroll
    for (int e = 0; e < DIM / 256; ++e) acc[e] += v * wrow[tid + e * 256];
  }
  float* orow = out + (size_t)row * DIM;
#pragma unroll
  for (int e = 0; e < DIM / 256; ++e) orow[tid + e * 256] = acc[e];
}

// ---------------- launcher ----------------
extern "C" void kernel_launch(void* const* d_in, const int* in_sizes, int n_in,
                              void* d_out, int out_size, void* d_ws, size_t ws_size,
                              hipStream_t stream) {
  const float* x      = (const float*)d_in[0];
  const float* W_enc  = (const float*)d_in[1];
  const float* W_dec  = (const float*)d_in[2];
  const float* in_b   = (const float*)d_in[3];
  const float* neur_b = (const float*)d_in[4];
  const int*   steps  = (const int*)  d_in[5];

  char* ws = (char*)d_ws;
  __bf16* wsA   = (__bf16*)(ws + OFF_A);
  __bf16* wsW   = (__bf16*)(ws + OFF_W);
  float*  wsPre = (float*) (ws + OFF_PRE);
  float*  wsWt  = (float*) (ws + OFF_WT);
  float*  wsMkV = (float*) (ws + OFF_MKV);
  int*    wsMkI = (int*)   (ws + OFF_MKI);
  float*  wsTkV = (float*) (ws + OFF_TKV);
  int*    wsTkI = (int*)   (ws + OFF_TKI);
  int*    wsSt  = (int*)   (ws + OFF_ST);

  float* out    = (float*)d_out;
  float* o_rec  = out;
  float* o_act  = o_rec  + (size_t)BATCH * DIM;
  float* o_tki  = o_act  + (size_t)BATCH * NEUR;
  float* o_tkv  = o_tki  + (size_t)BATCH * TOPK;
  float* o_mkr  = o_tkv  + (size_t)BATCH * TOPK;
  float* o_auxi = o_mkr  + (size_t)BATCH * DIM;
  float* o_auxv = o_auxi + (size_t)BATCH * AUXK;

  k_convert_x<<<(BATCH * DIM) / (256 * 8), 256, 0, stream>>>(x, in_b, wsA);
  k_convert_w<<<(NEUR * DIM) / (256 * 8), 256, 0, stream>>>(W_enc, wsW);
  k_transpose<<<dim3(NEUR / 32, DIM / 32), dim3(32, 8), 0, stream>>>(W_dec, wsWt);
  k_gemm<<<dim3(NEUR / 128, BATCH / 128), 256, 0, stream>>>(wsA, wsW, neur_b, wsPre);
  k_steps_init<<<NEUR / 256, 256, 0, stream>>>(steps, wsSt);
  k_topk<<<BATCH, 256, 0, stream>>>(wsPre, wsTkI, wsTkV, wsMkI, wsMkV, wsSt,
                                    o_act, o_tki, o_tkv);
  k_aux<<<BATCH, 256, 0, stream>>>(wsPre, wsSt, o_auxi, o_auxv);
  k_recon<<<BATCH, 256, 0, stream>>>(wsTkV, wsTkI, TOPK,   wsWt, in_b, o_rec);
  k_recon<<<BATCH, 256, 0, stream>>>(wsMkV, wsMkI, MULTIK, wsWt, in_b, o_mkr);
}
